// GRU2D_18511309045776
// MI455X (gfx1250) — matrix operand order repaired and checked by reading, not compiled, not run on hardware
//
#include <hip/hip_runtime.h>

// ---------------------------------------------------------------------------
// GRU2D on MI455X (gfx1250):
//   * rank-1 collapse of the input GEMM: a_g[t] = x[t]*(We@Wg) + (be@Wg + b_g)
//   * 1024-step recurrence: bf16 weights (L2-resident, 24 MB) streamed into
//     V_WMMA_F32_16X16X32_BF16 mat-vecs (h in A row 0), fp32 accumulate.
//   * Non-carrier A lanes read a zeroed LDS pad -> no per-iteration cndmask
//     masking, no WMMA->VALU hazard NOPs in the hot loop.
// ---------------------------------------------------------------------------

#define H 2048
#define NSTEP 1024
#define NT (H / 16)       // 128 column tiles per gate
#define KB (H / 32)       // 64 k-blocks per matvec
#define STEP_BLOCKS 16
#define STEP_THREADS 768  // 24 waves: 3 gates x 8 tiles

typedef __bf16        bf16x16 __attribute__((ext_vector_type(16)));
typedef float         v8f     __attribute__((ext_vector_type(8)));
typedef unsigned int  uint4v  __attribute__((ext_vector_type(4)));

union Frag {
    uint4v  u[2];
    bf16x16 v;
};

static __device__ __forceinline__ unsigned short f2bf(float f) {
    unsigned int u = __builtin_bit_cast(unsigned int, f);
    unsigned int lsb = (u >> 16) & 1u;
    u += 0x7FFFu + lsb;               // round-to-nearest-even
    return (unsigned short)(u >> 16);
}

// ---------------------------------------------------------------------------
// Pack Whr/Whz/Whn (fp32 row-major [k][j]) into bf16 B-fragment layout.
// B 32x16 bf16 fragment: lane holds col = tile*16 + (lane&15),
//   K = kblk*32 + (lane<16 ? 0 : 16) + e  for e in [0,16)  -> 16 contiguous
//   bf16 per (gate, tile, kblk, lane).
// ---------------------------------------------------------------------------
__global__ void pack_wh_kernel(const float* __restrict__ Whr,
                               const float* __restrict__ Whz,
                               const float* __restrict__ Whn,
                               unsigned short* __restrict__ wpack) {
    int i = blockIdx.x * blockDim.x + threadIdx.x;   // 3*128*64*32 threads
    int lane = i & 31;
    int kb   = (i >> 5) & 63;
    int nt   = (i >> 11) & 127;
    int g    = i >> 18;
    const float* W = (g == 0) ? Whr : (g == 1) ? Whz : Whn;
    int c     = nt * 16 + (lane & 15);
    int kbase = kb * 32 + ((lane < 16) ? 0 : 16);
    unsigned short* out = wpack + (size_t)i * 16;
#pragma unroll
    for (int e = 0; e < 16; ++e) {
        out[e] = f2bf(W[(size_t)(kbase + e) * H + c]);
    }
}

// ---------------------------------------------------------------------------
// One-time rank-1 collapse: Weg[g][j] = sum_k We[k]*Wg[k][j]
//                           beg[g][j] = sum_k be[k]*Wg[k][j] + b_g[j]
// ---------------------------------------------------------------------------
__global__ void weg_kernel(const float* __restrict__ We, const float* __restrict__ be,
                           const float* __restrict__ Wir, const float* __restrict__ Wiz,
                           const float* __restrict__ Win,
                           const float* __restrict__ bir, const float* __restrict__ biz,
                           const float* __restrict__ bin_,
                           float* __restrict__ Weg, float* __restrict__ beg) {
    int j  = blockIdx.x * blockDim.x + threadIdx.x;   // 0 .. 3*H-1
    int g  = j >> 11;
    int jj = j & (H - 1);
    const float* W  = (g == 0) ? Wir : (g == 1) ? Wiz : Win;
    const float* bg = (g == 0) ? bir : (g == 1) ? biz : bin_;
    float aw = 0.f, ab = 0.f;
    for (int k = 0; k < H; ++k) {
        float w = W[(size_t)k * H + jj];
        aw = fmaf(We[k], w, aw);
        ab = fmaf(be[k], w, ab);
    }
    Weg[j] = aw;
    beg[j] = ab + bg[jj];
}

__global__ void init_kernel(float* __restrict__ h,
                            unsigned short* __restrict__ hbf0,
                            unsigned short* __restrict__ hbf1) {
    int j = blockIdx.x * blockDim.x + threadIdx.x;
    if (j < H) { h[j] = 0.f; hbf0[j] = 0; hbf1[j] = 0; }
}

// ---------------------------------------------------------------------------
// One GRU step. Block b owns columns [128b, 128b+128) for all three gates:
// 24 waves each do a 1x2048 @ 2048x16 matvec via WMMA (h in A row 0),
// then threads 0..127 apply the gate math and update h (fp32 in d_out)
// plus the double-buffered bf16 copy for the next step's A fragments.
// ---------------------------------------------------------------------------
__global__ __launch_bounds__(STEP_THREADS, 1)
void gru_step_kernel(const float* __restrict__ x,
                     const float* __restrict__ Weg,
                     const float* __restrict__ beg,
                     const float* __restrict__ bhn,
                     const unsigned short* __restrict__ wpack,
                     const unsigned short* __restrict__ hbf_in,
                     unsigned short* __restrict__ hbf_out,
                     float* __restrict__ h,
                     int t) {
    // h in bf16 (4 KB) + 32-element zero pad for non-carrier A lanes
    __shared__ __align__(16) unsigned short hs[H + 32];
    __shared__ float u_s[3][128];         // per-gate matvec results for slice

    const int tid  = threadIdx.x;
    const int lane = tid & 31;
    const int wave = tid >> 5;            // 0..23
    const int b    = blockIdx.x;          // 0..15

    // Stage h_bf16 into LDS once (read as dwords: 1024 x b32) + zero the pad.
    for (int i = tid; i < H / 2; i += STEP_THREADS)
        ((unsigned int*)hs)[i] = ((const unsigned int*)hbf_in)[i];
    if (tid < 16) ((unsigned int*)hs)[H / 2 + tid] = 0u;
    __syncthreads();

    // ---- WMMA mat-vec: gate g, column tile nt ----
    const int g  = wave >> 3;             // 0:r 1:z 2:n
    const int nt = b * 8 + (wave & 7);    // global 16-col tile
    const unsigned short* wp =
        wpack + ((size_t)(g * NT + nt) * KB * 512) + (size_t)lane * 16;

    // A row-0 carriers are lanes 0 and 16; every other lane permanently
    // reads the 32-entry zero pad (loop-invariant base, zero stride).
    const bool keep  = (lane == 0) || (lane == 16);
    int        abase = keep ? ((lane & 16) ? 16 : 0) : H;
    const int  astep = keep ? 32 : 0;

    v8f c = {};
    for (int kb = 0; kb < KB; ++kb) {
        Frag a, bf;
        a.u[0] = *(const uint4v*)(hs + abase);        // e = 0..7
        a.u[1] = *(const uint4v*)(hs + abase + 8);    // e = 8..15
        const uint4v* bp = (const uint4v*)(wp + (size_t)kb * 512);
        bf.u[0] = bp[0];
        bf.u[1] = bp[1];
        c = __builtin_amdgcn_wmma_f32_16x16x32_bf16(
                false, a.v, false, bf.v, (short)0, c, false, false);
        abase += astep;
    }
    // Row 0 of D lives in VGPR0 (c[0]) of lanes 0..15.
    if (lane < 16) u_s[g][(wave & 7) * 16 + lane] = c[0];
    __syncthreads();

    // ---- gate math + h update for this block's 128 columns ----
    if (tid < 128) {
        const int j = b * 128 + tid;
        // snake order: row-major with odd rows reversed
        const int row  = t >> 5, col = t & 31;
        const int xi   = row * 32 + ((row & 1) ? (31 - col) : col);
        const float xt = x[xi];

        const float ar = fmaf(xt, Weg[0 * H + j], beg[0 * H + j]);
        const float az = fmaf(xt, Weg[1 * H + j], beg[1 * H + j]);
        const float an = fmaf(xt, Weg[2 * H + j], beg[2 * H + j]);

        const float r = 1.0f / (1.0f + __expf(-(ar + u_s[0][tid])));
        const float z = 1.0f / (1.0f + __expf(-(az + u_s[1][tid])));
        const float n = tanhf(an + r * (u_s[2][tid] + bhn[j]));

        const float hn = (1.0f - z) * n + z * h[j];
        h[j]       = hn;          // fp32 state (d_out) — block-private slice
        hbf_out[j] = f2bf(hn);    // bf16 copy for next step's A fragments
    }
}

// ---------------------------------------------------------------------------
extern "C" void kernel_launch(void* const* d_in, const int* in_sizes, int n_in,
                              void* d_out, int out_size, void* d_ws, size_t ws_size,
                              hipStream_t stream) {
    (void)in_sizes; (void)n_in; (void)out_size; (void)ws_size;

    const float* x    = (const float*)d_in[0];
    const float* We   = (const float*)d_in[1];
    const float* be   = (const float*)d_in[2];
    const float* Wir  = (const float*)d_in[3];
    const float* bir  = (const float*)d_in[4];
    const float* Wiz  = (const float*)d_in[5];
    const float* biz  = (const float*)d_in[6];
    const float* Win  = (const float*)d_in[7];
    const float* bin_ = (const float*)d_in[8];
    const float* Whr  = (const float*)d_in[9];
    const float* Whz  = (const float*)d_in[10];
    const float* Whn  = (const float*)d_in[11];
    const float* bhn  = (const float*)d_in[12];

    float* h = (float*)d_out;   // h state lives directly in the output buffer

    // Workspace layout
    char* ws = (char*)d_ws;
    unsigned short* wpack = (unsigned short*)ws;                    // 3*H*H bf16 = 24 MB
    size_t off = (size_t)3 * H * H * sizeof(unsigned short);
    unsigned short* hbf0 = (unsigned short*)(ws + off); off += H * sizeof(unsigned short) * 2;
    unsigned short* hbf1 = hbf0 + H;
    float* Weg = (float*)(ws + off); off += 3 * H * sizeof(float);
    float* beg = (float*)(ws + off);

    // One-time prep (HBM -> L2-resident bf16 weights; rank-1 collapse)
    pack_wh_kernel<<<(3 * NT * KB * 32) / 256, 256, 0, stream>>>(Whr, Whz, Whn, wpack);
    weg_kernel<<<(3 * H) / 256, 256, 0, stream>>>(We, be, Wir, Wiz, Win,
                                                  bir, biz, bin_, Weg, beg);
    init_kernel<<<(H + 255) / 256, 256, 0, stream>>>(h, hbf0, hbf1);

    // Sequential scan: kernel boundary = inter-step barrier (graph-safe).
    for (int t = 0; t < NSTEP; ++t) {
        const unsigned short* hin  = (t & 1) ? hbf1 : hbf0;
        unsigned short*       hout = (t & 1) ? hbf0 : hbf1;
        gru_step_kernel<<<STEP_BLOCKS, STEP_THREADS, 0, stream>>>(
            x, Weg, beg, bhn, wpack, hin, hout, h, t);
    }
}
